// EdgeUpdate_53730040873192
// MI455X (gfx1250) — matrix operand order repaired and checked by reading, not compiled, not run on hardware
//
#include <hip/hip_runtime.h>

#define NODE_DIM 128
#define EDGE_DIMC 128
#define GLOB_DIM 64
#define NGRAPH 64
#define INDIM 448          // 2*128 + 128 + 64
#define HID 512
#define EPSV 1e-5f

#define MT 128             // edge rows per block (8 waves x 16 rows)
#define XA_STRIDE 456      // 448 + 8 pad (16B aligned rows, spread banks)
#define HB_STRIDE 72       // 64 + 8 pad
#define OUT_STRIDE 132     // fp32 out tile stride (bank-spread)
#define SPLITS 64          // row splits per graph in stats reduction

#define XA_ELEMS  (MT * XA_STRIDE)        // 58368
#define HB_ELEMS  (8 * 16 * HB_STRIDE)    // 9216
#define W1C_ELEMS (64 * XA_STRIDE)        // 29184 (64 cols x 448(+pad) K)
#define W2C_ELEMS (128 * HB_STRIDE)       // 9216  (128 cols x 64(+pad) K)
#define LDS_ELEMS (XA_ELEMS + HB_ELEMS + 2 * W1C_ELEMS + 2 * W2C_ELEMS) // 144384
// total LDS bytes = 288768  (<= 320KB per workgroup)

typedef __attribute__((ext_vector_type(16))) _Float16 v16h;
typedef __attribute__((ext_vector_type(8)))  float    v8f;

union Frag { v16h v; uint4 q[2]; };

// 16B global -> LDS async copy (per-lane), tracked by ASYNCcnt
__device__ __forceinline__ void async_cp16(const _Float16* l, const _Float16* g)
{
    unsigned ldsOff = (unsigned)(unsigned long long)l;   // low 32 bits = LDS byte offset
    asm volatile("global_load_async_to_lds_b128 %0, %1, off"
                 :: "v"(ldsOff), "v"(g)
                 : "memory");
}
__device__ __forceinline__ void wait_async0()
{
    asm volatile("s_wait_asynccnt 0x0" ::: "memory");
}

// ---------------------------------------------------------------- prep ----
__global__ __launch_bounds__(256)
void prep_kernel(const float* __restrict__ W1, const float* __restrict__ W2,
                 const int* __restrict__ batch, int E,
                 _Float16* __restrict__ w1t, _Float16* __restrict__ w2t,
                 float* __restrict__ S1, float* __restrict__ S2,
                 int* __restrict__ starts)
{
    const int NW1 = INDIM * HID;            // 229376
    const int NW2 = HID * EDGE_DIMC;        // 65536
    const int NZ  = 2 * NGRAPH * EDGE_DIMC; // 16384
    int i = blockIdx.x * 256 + threadIdx.x;
    if (i < NW1) {
        int n = i / INDIM, k = i - n * INDIM;        // W1T[n][k] = W1[k][n]
        w1t[n * INDIM + k] = (_Float16)W1[k * HID + n];
    } else if (i < NW1 + NW2) {
        int j = i - NW1;
        int n = j / HID, k = j - n * HID;            // W2T[n][k] = W2[k][n]
        w2t[n * HID + k] = (_Float16)W2[k * EDGE_DIMC + n];
    } else if (i < NW1 + NW2 + NZ) {
        int j = i - NW1 - NW2;
        if (j < NGRAPH * EDGE_DIMC) S1[j] = 0.f;
        else                        S2[j - NGRAPH * EDGE_DIMC] = 0.f;
    } else if (i < NW1 + NW2 + NZ + NGRAPH + 1) {
        int g = i - NW1 - NW2 - NZ;
        if (g == NGRAPH) { starts[g] = E; }
        else {
            int lo = 0, hi = E;                       // lower_bound(batch, g)
            while (lo < hi) { int m = (lo + hi) >> 1; if (batch[m] < g) lo = m + 1; else hi = m; }
            starts[g] = lo;
        }
    }
}

// --------------------------------------------------- async weight staging ----
__device__ __forceinline__ void stage_w_chunk(const _Float16* __restrict__ w1t,
                                              const _Float16* __restrict__ w2t,
                                              _Float16* w1c, _Float16* w2c,
                                              int hc, int t)
{
    // W1T slab: 64 cols x 448 K  -> 64 x 56 uint4 = 3584 copies (14 per thread)
#pragma unroll
    for (int i = 0; i < 14; ++i) {
        int idx  = t + i * 256;
        int col  = idx / 56;
        int part = idx - col * 56;
        async_cp16(w1c + col * XA_STRIDE + part * 8,
                   w1t + (hc * 64 + col) * INDIM + part * 8);
    }
    // W2T slab: 128 cols x 64 K -> 128 x 8 uint4 = 1024 copies (4 per thread)
#pragma unroll
    for (int i = 0; i < 4; ++i) {
        int idx  = t + i * 256;
        int col  = idx >> 3;
        int part = idx & 7;
        async_cp16(w2c + col * HB_STRIDE + part * 8,
                   w2t + col * HID + hc * 64 + part * 8);
    }
}

// ------------------------------------------------------- fused MLP (WMMA) ----
__global__ __launch_bounds__(256)
void mlp_kernel(const float* __restrict__ srcp, const float* __restrict__ dstp,
                const float* __restrict__ ea,   const float* __restrict__ u,
                const int* __restrict__ batch,
                const _Float16* __restrict__ w1t, const _Float16* __restrict__ w2t,
                const float* __restrict__ b1, const float* __restrict__ b2,
                float* __restrict__ xout, int E)
{
    extern __shared__ __attribute__((aligned(16))) _Float16 lds[];
    _Float16* xa   = lds;                          // [MT][XA_STRIDE] f16 concat
    _Float16* hb   = lds + XA_ELEMS;               // 8 waves x [16][HB_STRIDE]
    _Float16* w1c0 = lds + XA_ELEMS + HB_ELEMS;    // double-buffered W1T slab
    _Float16* w1c1 = w1c0 + W1C_ELEMS;
    _Float16* w2c0 = w1c1 + W1C_ELEMS;             // double-buffered W2T slab
    _Float16* w2c1 = w2c0 + W2C_ELEMS;

    const int t = threadIdx.x;
    const long long tileBase = (long long)blockIdx.x * MT;

    // kick off async staging of chunk 0 weights (overlaps xa staging below)
    stage_w_chunk(w1t, w2t, w1c0, w2c0, 0, t);

    // ---- stage concat tile to LDS as f16 (float4 loads, ushort4 stores)
    for (int idx = t; idx < MT * 112; idx += 256) {       // 112 float4 per row
        int r  = idx / 112;
        int c4 = idx - r * 112;
        long long R = tileBase + r; if (R >= E) R = E - 1;
        int c = c4 * 4;
        float4 f;
        if      (c < 128) f = *(const float4*)(srcp + R * 128 + c);
        else if (c < 256) f = *(const float4*)(dstp + R * 128 + (c - 128));
        else if (c < 384) f = *(const float4*)(ea   + R * 128 + (c - 256));
        else { int g = batch[R]; f = *(const float4*)(u + (long long)g * 64 + (c - 384)); }
        union { _Float16 h[4]; ushort4 u4; } pk;
        pk.h[0] = (_Float16)f.x; pk.h[1] = (_Float16)f.y;
        pk.h[2] = (_Float16)f.z; pk.h[3] = (_Float16)f.w;
        *(ushort4*)(xa + r * XA_STRIDE + c) = pk.u4;
    }
    wait_async0();
    __syncthreads();

    const int wv   = t >> 5;
    const int lane = t & 31;
    const int half = lane >> 4;
    const int ln   = lane & 15;

    const _Float16* arow = xa + (wv * 16 + ln) * XA_STRIDE;   // A rows (M = ln)
    _Float16* hrow = hb + wv * 16 * HB_STRIDE;                // this wave's h1 buf

    v8f zero = {0.f, 0.f, 0.f, 0.f, 0.f, 0.f, 0.f, 0.f};
    v8f acc2[8];
#pragma unroll
    for (int j = 0; j < 8; ++j) acc2[j] = zero;

    for (int hc = 0; hc < 8; ++hc) {                 // hidden chunks of 64
        const _Float16* w1c = (hc & 1) ? w1c1 : w1c0;
        const _Float16* w2c = (hc & 1) ? w2c1 : w2c0;
        if (hc + 1 < 8)                              // prefetch next chunk
            stage_w_chunk(w1t, w2t, (hc & 1) ? w1c0 : w1c1,
                          (hc & 1) ? w2c0 : w2c1, hc + 1, t);

        v8f acc1[4];
#pragma unroll
        for (int j = 0; j < 4; ++j) acc1[j] = zero;

        const _Float16* bbase = w1c + ln * XA_STRIDE + 16 * half;
#pragma unroll
        for (int k0 = 0; k0 < INDIM; k0 += 32) {     // GEMM1: K = 448
            // --- issue all loads for this k-step as one group ---
            Frag A, B[4];
            A.q[0] = *(const uint4*)(arow + k0 + 16 * half);      // e0..7
            A.q[1] = *(const uint4*)(arow + k0 + 16 * half + 8);  // e8..15
#pragma unroll
            for (int j = 0; j < 4; ++j) {
                const _Float16* bp = bbase + j * 16 * XA_STRIDE + k0;
                B[j].q[0] = *(const uint4*)(bp);
                B[j].q[1] = *(const uint4*)(bp + 8);
            }
            // --- then 4 independent WMMAs (one dscnt wait per group) ---
#pragma unroll
            for (int j = 0; j < 4; ++j)
                acc1[j] = __builtin_amdgcn_wmma_f32_16x16x32_f16(
                    false, A.v, false, B[j].v, (short)0, acc1[j], false, false);
        }

        // bias + ReLU -> per-wave LDS staging (re-fragment as A for GEMM2)
#pragma unroll
        for (int j = 0; j < 4; ++j) {
            float bv = b1[hc * 64 + j * 16 + ln];
#pragma unroll
            for (int e = 0; e < 8; ++e) {
                float v = acc1[j][e] + bv;
                v = v > 0.f ? v : 0.f;
                hrow[(e + 8 * half) * HB_STRIDE + j * 16 + ln] = (_Float16)v;
            }
        }
        // same-wave DS ops are in-order: no barrier needed before re-read

        const _Float16* h2p = hrow + ln * HB_STRIDE;
        const _Float16* b2base = w2c + ln * HB_STRIDE + 16 * half;
#pragma unroll
        for (int k2 = 0; k2 < 64; k2 += 32) {        // GEMM2 over this chunk
            Frag A2;
            A2.q[0] = *(const uint4*)(h2p + k2 + 16 * half);
            A2.q[1] = *(const uint4*)(h2p + k2 + 16 * half + 8);
#pragma unroll
            for (int jh = 0; jh < 2; ++jh) {         // two groups of 4
                Frag B2[4];
#pragma unroll
                for (int j = 0; j < 4; ++j) {
                    const _Float16* bp = b2base + (jh * 4 + j) * 16 * HB_STRIDE + k2;
                    B2[j].q[0] = *(const uint4*)(bp);
                    B2[j].q[1] = *(const uint4*)(bp + 8);
                }
#pragma unroll
                for (int j = 0; j < 4; ++j)
                    acc2[jh * 4 + j] = __builtin_amdgcn_wmma_f32_16x16x32_f16(
                        false, A2.v, false, B2[j].v, (short)0, acc2[jh * 4 + j],
                        false, false);
            }
        }

        wait_async0();       // next chunk's weights landed in LDS
        __syncthreads();     // also fences buffer reuse across waves
    }

    // ---- epilogue: dump tile to LDS (xa is dead), then coalesced writeout
    float* outb = (float*)lds;                       // [MT][OUT_STRIDE] fp32
#pragma unroll
    for (int jn = 0; jn < 8; ++jn) {
        int n = jn * 16 + ln;
        float bv = b2[n];
#pragma unroll
        for (int e = 0; e < 8; ++e)
            outb[(wv * 16 + e + 8 * half) * OUT_STRIDE + n] = acc2[jn][e] + bv;
    }
    __syncthreads();

    for (int idx = t; idx < MT * 32; idx += 256) {   // 32 float4 per row
        int r = idx >> 5;
        int c = (idx & 31) * 4;
        long long R = tileBase + r;
        if (R < E) {
            float4 va = *(const float4*)(outb + r * OUT_STRIDE + c);
            float4 fe = *(const float4*)(ea + R * 128 + c);   // exact fp32 residual
            va.x += fe.x; va.y += fe.y; va.z += fe.z; va.w += fe.w;
            *(float4*)(xout + R * 128 + c) = va;
        }
    }
}

// -------------------------------------------------- GraphNorm statistics ----
__global__ __launch_bounds__(128)
void stats_part(const float* __restrict__ x, const int* __restrict__ starts,
                float* __restrict__ S1, float* __restrict__ S2)
{
    int g  = blockIdx.x / SPLITS;
    int sp = blockIdx.x - g * SPLITS;
    int c  = threadIdx.x;                 // 0..127
    int lo = starts[g], hi = starts[g + 1];
    float s1 = 0.f, s2 = 0.f;
    for (long long r = lo + sp; r < hi; r += SPLITS) {
        float v = x[r * 128 + c];
        s1 += v; s2 += v * v;
    }
    atomicAdd(&S1[g * 128 + c], s1);
    atomicAdd(&S2[g * 128 + c], s2);
}

__global__ __launch_bounds__(128)
void stats_final(const float* __restrict__ S1, const float* __restrict__ S2,
                 const int* __restrict__ starts,
                 const float* __restrict__ gw, const float* __restrict__ gb,
                 const float* __restrict__ ms,
                 float* __restrict__ alpha, float* __restrict__ beta)
{
    int i = blockIdx.x * 128 + threadIdx.x;      // 64*128
    int g = i >> 7, c = i & 127;
    float cnt = (float)(starts[g + 1] - starts[g]);
    if (cnt < 1.f) cnt = 1.f;
    float mean = S1[i] / cnt;
    float m2   = mean * ms[c];                   // mean * mean_scale
    // var(x - m2) = E[x^2] - 2*m2*E[x] + m2^2
    float var  = S2[i] / cnt - 2.f * m2 * mean + m2 * m2;
    float inv  = rsqrtf(var + EPSV);
    float a    = inv * gw[c];
    alpha[i] = a;
    beta[i]  = gb[c] - m2 * a;
}

// -------------------------------------------------------------- finalize ----
__global__ __launch_bounds__(256)
void finalize_kernel(const float* __restrict__ x, const int* __restrict__ batch,
                     const float* __restrict__ alpha, const float* __restrict__ beta,
                     float* __restrict__ out, long long total4)
{
    long long i = (long long)blockIdx.x * 256 + threadIdx.x;
    if (i >= total4) return;
    long long r = i >> 5;                        // 32 float4 per row
    int c = (int)(i & 31) * 4;
    int g = batch[r];
    float4 xv = *(const float4*)(x + r * 128 + c);
    const float* ap = alpha + g * 128 + c;
    const float* bp = beta  + g * 128 + c;
    float4 o;
    o.x = xv.x * ap[0] + bp[0];
    o.y = xv.y * ap[1] + bp[1];
    o.z = xv.z * ap[2] + bp[2];
    o.w = xv.w * ap[3] + bp[3];
    *(float4*)(out + r * 128 + c) = o;
}

// ---------------------------------------------------------------- launch ----
extern "C" void kernel_launch(void* const* d_in, const int* in_sizes, int n_in,
                              void* d_out, int out_size, void* d_ws, size_t ws_size,
                              hipStream_t stream)
{
    const float* srcp  = (const float*)d_in[0];
    const float* dstp  = (const float*)d_in[1];
    const float* ea    = (const float*)d_in[2];
    const float* u     = (const float*)d_in[3];
    const int*   batch = (const int*)d_in[4];
    const float* W1    = (const float*)d_in[5];
    const float* b1    = (const float*)d_in[6];
    const float* W2    = (const float*)d_in[7];
    const float* b2    = (const float*)d_in[8];
    const float* gw    = (const float*)d_in[9];
    const float* gb    = (const float*)d_in[10];
    const float* ms    = (const float*)d_in[11];
    const int E = in_sizes[0] / NODE_DIM;

    char* ws = (char*)d_ws;
    size_t off = 0;
    float* x = (float*)(ws + off);       off += (size_t)E * 128 * 4;
    off = (off + 255) & ~(size_t)255;
    _Float16* w1t = (_Float16*)(ws + off); off += (size_t)HID * INDIM * 2;
    _Float16* w2t = (_Float16*)(ws + off); off += (size_t)EDGE_DIMC * HID * 2;
    float* S1    = (float*)(ws + off);   off += NGRAPH * 128 * 4;
    float* S2    = (float*)(ws + off);   off += NGRAPH * 128 * 4;
    float* alpha = (float*)(ws + off);   off += NGRAPH * 128 * 4;
    float* beta  = (float*)(ws + off);   off += NGRAPH * 128 * 4;
    int* starts  = (int*)(ws + off);     off += (NGRAPH + 1) * 4;

    const int prepN = INDIM * HID + HID * EDGE_DIMC + 2 * NGRAPH * 128 + NGRAPH + 1;
    prep_kernel<<<(prepN + 255) / 256, 256, 0, stream>>>(
        W1, W2, batch, E, w1t, w2t, S1, S2, starts);

    const int tiles = (E + MT - 1) / MT;
    const size_t ldsBytes = (size_t)LDS_ELEMS * sizeof(_Float16);   // 288768
    (void)hipFuncSetAttribute((const void*)mlp_kernel,
                              hipFuncAttributeMaxDynamicSharedMemorySize,
                              (int)ldsBytes);
    mlp_kernel<<<tiles, 256, ldsBytes, stream>>>(
        srcp, dstp, ea, u, batch, w1t, w2t, b1, b2, x, E);

    stats_part<<<NGRAPH * SPLITS, 128, 0, stream>>>(x, starts, S1, S2);
    stats_final<<<NGRAPH, 128, 0, stream>>>(S1, S2, starts, gw, gb, ms, alpha, beta);

    const long long total4 = (long long)E * 32;
    finalize_kernel<<<(int)((total4 + 255) / 256), 256, 0, stream>>>(
        x, batch, alpha, beta, (float*)d_out, total4);
}
